// PointTransformerModel_48790828483150
// MI455X (gfx1250) — matrix operand order repaired and checked by reading, not compiled, not run on hardware
//
#include <hip/hip_runtime.h>
#include <hip/hip_bf16.h>
#include <math.h>

// ---------------------------------------------------------------------------
// CDNA5 (gfx1250) wave32 WMMA types / helpers
// ---------------------------------------------------------------------------
typedef __attribute__((ext_vector_type(16))) __bf16        v16bf;
typedef __attribute__((ext_vector_type(8)))  float         v8f;
typedef __attribute__((ext_vector_type(8)))  unsigned int  v8u;

__device__ __forceinline__ v8f wmma_bf16(const v16bf a, const v16bf b, const v8f c) {
    // (neg_a, A, neg_b, B, c_mod, C, reuse_a, reuse_b)
    return __builtin_amdgcn_wmma_f32_16x16x32_bf16(false, a, false, b, (short)0, c,
                                                   false, false);
}

// pack two f32 -> one dword of two bf16 (lowers to v_cvt_pk_bf16_f32)
__device__ __forceinline__ unsigned int pack_bf16(float a, float b) {
    unsigned short lo = __builtin_bit_cast(unsigned short, (__bf16)a);
    unsigned short hi = __builtin_bit_cast(unsigned short, (__bf16)b);
    return (unsigned int)lo | ((unsigned int)hi << 16);
}

// A-fragment (16xK bf16, per-lane row) from a packed-dword LDS row.
// element j <-> K = j + 8*half + (j>=8 ? 8 : 0)  =>  dword d -> d + 4*half + (d>=4?4:0)
__device__ __forceinline__ v16bf fragA(const unsigned int* row, int half, int base) {
    v8u u;
#pragma unroll
    for (int d = 0; d < 8; ++d)
        u[d] = row[base + d + 4 * half + ((d >= 4) ? 4 : 0)];
    return __builtin_bit_cast(v16bf, u);
}

// B-fragment (Kx16 bf16) from an N-major packed-dword LDS row (lane's column).
// element j <-> K = j + 8*half + (j>=8 ? 8 : 0)  =>  dword d -> d + 4*half + (d>=4?4:0)
__device__ __forceinline__ v16bf fragB(const unsigned int* col, int half, int base) {
    v8u u;
#pragma unroll
    for (int d = 0; d < 8; ++d)
        u[d] = col[base + d + 4 * half + ((d >= 4) ? 4 : 0)];
    return __builtin_bit_cast(v16bf, u);
}

// ---------------------------------------------------------------------------
// Generic GEMM:  C[M,N] = act(A[M,K] @ W[K,N] + bias (+ resid))
// 128 threads (4 waves), block tile 64x16, one 16x16 WMMA tile per wave.
// LDS tiles stored as packed bf16 dwords; W tile stored N-major so each
// lane's B column is K-contiguous (b32/b64 LDS fragment loads).
// ---------------------------------------------------------------------------
__global__ __launch_bounds__(128)
void gemm_bf16_kernel(const float* __restrict__ A, const float* __restrict__ W,
                      const float* __restrict__ bias, const float* __restrict__ resid,
                      float* __restrict__ C, int M, int N, int Kdim, int relu)
{
    __shared__ unsigned int As32[64][17];   // 64 rows x 16 K-dwords (padded)
    __shared__ unsigned int Wt32[16][18];   // 16 cols x 16 K-dwords (padded)

    const int tid  = threadIdx.x;
    const int lane = tid & 31;
    const int wv   = tid >> 5;
    const int half = lane >> 4;
    const int ml   = lane & 15;

    const int mBase = blockIdx.x * 64;
    const int nBase = blockIdx.y * 16;

    v8f acc = {};

    for (int k0 = 0; k0 < Kdim; k0 += 32) {
        const bool fastA = ((Kdim & 3) == 0) && (mBase + 64 <= M) && (k0 + 32 <= Kdim);
        if (fastA) {
            // full interior tile: float4 loads (global_load_b128) + packed stores
            for (int t = tid; t < 64 * 8; t += 128) {
                int r = t >> 3, d4 = t & 7;
                const float4* p =
                    reinterpret_cast<const float4*>(A + (size_t)(mBase + r) * Kdim + k0);
                float4 v = p[d4];
                As32[r][d4 * 2 + 0] = pack_bf16(v.x, v.y);
                As32[r][d4 * 2 + 1] = pack_bf16(v.z, v.w);
            }
        } else {
            for (int t = tid; t < 64 * 16; t += 128) {
                int r = t >> 4, c2 = t & 15;
                int gm = mBase + r, gk = k0 + c2 * 2;
                float a0 = (gm < M && gk     < Kdim) ? A[(size_t)gm * Kdim + gk]     : 0.f;
                float a1 = (gm < M && gk + 1 < Kdim) ? A[(size_t)gm * Kdim + gk + 1] : 0.f;
                As32[r][c2] = pack_bf16(a0, a1);
            }
        }
        // W tile (32x16) stored N-major as packed K-pairs
        for (int t = tid; t < 256; t += 128) {
            int c = t & 15, r2 = t >> 4;
            int gn = nBase + c, gk = k0 + r2 * 2;
            float w0 = (gk     < Kdim && gn < N) ? W[(size_t)gk * N + gn]       : 0.f;
            float w1 = (gk + 1 < Kdim && gn < N) ? W[(size_t)(gk + 1) * N + gn] : 0.f;
            Wt32[c][r2] = pack_bf16(w0, w1);
        }
        // prefetch next K slab of A into L2 (global_prefetch_b8)
        if (k0 + 32 < Kdim && tid < 64) {
            int gm = mBase + tid;
            if (gm < M) __builtin_prefetch(&A[(size_t)gm * Kdim + k0 + 32], 0, 1);
        }
        __syncthreads();

        v16bf af  = fragA(&As32[wv * 16 + ml][0], half, 0);
        v16bf bf_ = fragB(&Wt32[ml][0], half, 0);
        acc = wmma_bf16(af, bf_, acc);
        __syncthreads();
    }

    // epilogue: C/D layout -> VGPR r holds M = r + 8*half, N = ml
    const int col = nBase + ml;
    if (col < N) {
        float bv = bias ? bias[col] : 0.f;
#pragma unroll
        for (int r = 0; r < 8; ++r) {
            int row = mBase + wv * 16 + r + 8 * half;
            if (row < M) {
                float v = acc[r] + bv;
                if (resid) v += resid[(size_t)row * N + col];
                if (relu)  v = fmaxf(v, 0.f);
                C[(size_t)row * N + col] = v;
            }
        }
    }
}

// ---------------------------------------------------------------------------
// Fused Point-Transformer attention core.  One wave per query point n:
// a 16-neighbor x 64-feature tile (exactly the WMMA M=16 shape).
//   pos    = delta1(relu(delta0(rel)))                 (WMMA x8)
//   t      = q - k[idx] + pos                          (VALU + gather)
//   logits = gamma1(relu(gamma0(t)))                   (WMMA x16)
//   attn   = softmax_k(logits / 8);  res = sum_k attn * (v[idx] + pos)
// Weights staged once per block in LDS (bf16, N-major, packed dwords);
// all (16,64) intermediates live in LDS -- nothing (B,N,16,64) hits HBM.
// ---------------------------------------------------------------------------
__global__ __launch_bounds__(128)
void pt_attn_kernel(const float* __restrict__ xyz, const int* __restrict__ knn,
                    const float* __restrict__ q,  const float* __restrict__ kf,
                    const float* __restrict__ vf,
                    const float* __restrict__ d0w, const float* __restrict__ d0b,
                    const float* __restrict__ d1w, const float* __restrict__ d1b,
                    const float* __restrict__ g0w, const float* __restrict__ g0b,
                    const float* __restrict__ g1w, const float* __restrict__ g1b,
                    float* __restrict__ res, int N, int kk)
{
    __shared__ unsigned int Wd1t[64][33];   // N-major packed K-pairs (K=64 -> 32 dw)
    __shared__ unsigned int Wg0t[64][33];
    __shared__ unsigned int Wg1t[64][33];
    __shared__ float  Wd0[3][64];
    __shared__ float  Bd0[64], Bd1[64], Bg0[64], Bg1[64];
    __shared__ unsigned int Af32[4][16][33]; // per-wave A tile, packed (reused per stage)
    __shared__ __bf16 posS [4][16][66];      // pos tile (bf16)
    __shared__ float  logitS[4][16][66];     // logits / attn
    __shared__ float  relS[4][16][3];
    __shared__ float  qS[4][64];
    __shared__ int    nidx[4][16];

    const int tid  = threadIdx.x;
    const int lane = tid & 31;
    const int wv   = tid >> 5;
    const int half = lane >> 4;
    const int ml   = lane & 15;
    const int b    = blockIdx.y;

    int n = blockIdx.x * 4 + wv;
    if (n >= N) n = N - 1;            // keep all waves active (EXEC all-ones for WMMA)

    // stage weights N-major, packed bf16 pairs along K
    for (int t = tid; t < 2048; t += 128) {
        int c = t >> 5, dk = t & 31;
        int k = dk * 2;
        Wd1t[c][dk] = pack_bf16(d1w[(size_t)k * 64 + c], d1w[(size_t)(k + 1) * 64 + c]);
        Wg0t[c][dk] = pack_bf16(g0w[(size_t)k * 64 + c], g0w[(size_t)(k + 1) * 64 + c]);
        Wg1t[c][dk] = pack_bf16(g1w[(size_t)k * 64 + c], g1w[(size_t)(k + 1) * 64 + c]);
    }
    for (int t = tid; t < 64; t += 128) {
        Bd0[t] = d0b[t]; Bd1[t] = d1b[t]; Bg0[t] = g0b[t]; Bg1[t] = g1b[t];
    }
    for (int t = tid; t < 192; t += 128) Wd0[t / 64][t & 63] = d0w[t];

    const float* xyzb = xyz + (size_t)b * N * 3;
    if (lane < 16) {
        int slot = (lane < kk) ? lane : 0;
        int j = knn[((size_t)b * N + n) * 16 + slot];
        nidx[wv][lane] = j;
        for (int d = 0; d < 3; ++d)
            relS[wv][lane][d] = (lane < kk) ? (xyzb[n * 3 + d] - xyzb[j * 3 + d]) : 0.f;
    }
    for (int c = lane; c < 64; c += 32) qS[wv][c] = q[((size_t)b * N + n) * 64 + c];
    __syncthreads();

    // delta0 (K=3) + relu -> Afrag (packed pairs)
    for (int t = lane; t < 512; t += 32) {
        int k = t >> 5, d = t & 31;
        int c = d * 2;
        float r0 = relS[wv][k][0], r1 = relS[wv][k][1], r2 = relS[wv][k][2];
        float v0 = Bd0[c]   + r0 * Wd0[0][c]   + r1 * Wd0[1][c]   + r2 * Wd0[2][c];
        float v1 = Bd0[c+1] + r0 * Wd0[0][c+1] + r1 * Wd0[1][c+1] + r2 * Wd0[2][c+1];
        Af32[wv][k][d] = pack_bf16(fmaxf(v0, 0.f), fmaxf(v1, 0.f));
    }
    __syncthreads();

    // ---- wave (16x64) @ (64x64) from LDS, 8 WMMAs ----
    auto wave_gemm = [&](const unsigned int (*Wt)[33], v8f out[4]) {
        v16bf a0 = fragA(&Af32[wv][ml][0], half, 0);
        v16bf a1 = fragA(&Af32[wv][ml][0], half, 16);
#pragma unroll
        for (int nt = 0; nt < 4; ++nt) {
            const unsigned int* col = &Wt[nt * 16 + ml][0];
            v16bf b0 = fragB(col, half, 0);
            v16bf b1 = fragB(col, half, 16);
            v8f acc = {};
            acc = wmma_bf16(a0, b0, acc);
            acc = wmma_bf16(a1, b1, acc);
            out[nt] = acc;
        }
    };

    v8f acc[4];

    // delta1 -> pos (bf16 in LDS)
    wave_gemm(Wd1t, acc);
#pragma unroll
    for (int nt = 0; nt < 4; ++nt) {
        int c = nt * 16 + ml; float bv = Bd1[c];
#pragma unroll
        for (int r = 0; r < 8; ++r) {
            int m = r + 8 * half;
            posS[wv][m][c] = (__bf16)(acc[nt][r] + bv);
        }
    }
    __syncthreads();

    // t = q - k_gather + pos -> Afrag (float2 gathers, packed stores)
    for (int t = lane; t < 512; t += 32) {
        int k = t >> 5, d = t & 31;
        int c = d * 2;
        int j = nidx[wv][k];
        const float2* kr = reinterpret_cast<const float2*>(kf + ((size_t)b * N + j) * 64);
        float2 kv = kr[d];
        float v0 = qS[wv][c]     - kv.x + (float)posS[wv][k][c];
        float v1 = qS[wv][c + 1] - kv.y + (float)posS[wv][k][c + 1];
        Af32[wv][k][d] = pack_bf16(v0, v1);
    }
    __syncthreads();

    // gamma0 + relu -> Afrag (in place: fragments pulled to VGPRs first)
    wave_gemm(Wg0t, acc);
    __syncthreads();
#pragma unroll
    for (int nt = 0; nt < 2; ++nt) {       // pack two adjacent 16-col tiles? no: keep per-col
        ;
    }
#pragma unroll
    for (int nt = 0; nt < 4; ++nt) {
        int c = nt * 16 + ml; float bv = Bg0[c];
#pragma unroll
        for (int r = 0; r < 8; ++r) {
            int m = r + 8 * half;
            // per-element bf16 store into packed dword (write halves via bf16 view)
            reinterpret_cast<__bf16*>(&Af32[wv][m][0])[c] =
                (__bf16)fmaxf(acc[nt][r] + bv, 0.f);
        }
    }
    __syncthreads();

    // gamma1 -> logits/8 in LDS (mask rows >= kk)
    wave_gemm(Wg1t, acc);
#pragma unroll
    for (int nt = 0; nt < 4; ++nt) {
        int c = nt * 16 + ml; float bv = Bg1[c];
#pragma unroll
        for (int r = 0; r < 8; ++r) {
            int m = r + 8 * half;
            logitS[wv][m][c] = (m < kk) ? (acc[nt][r] + bv) * 0.125f : -3.0e38f;
        }
    }
    __syncthreads();

    // v + pos -> Afrag (packed, masked)
    for (int t = lane; t < 512; t += 32) {
        int k = t >> 5, d = t & 31;
        int c = d * 2;
        int j = nidx[wv][k];
        const float2* vr = reinterpret_cast<const float2*>(vf + ((size_t)b * N + j) * 64);
        float2 vv = vr[d];
        float v0 = (k < kk) ? (vv.x + (float)posS[wv][k][c])     : 0.f;
        float v1 = (k < kk) ? (vv.y + (float)posS[wv][k][c + 1]) : 0.f;
        Af32[wv][k][d] = pack_bf16(v0, v1);
    }
    __syncthreads();

    // softmax over k (per feature column) + weighted sum -> res
    for (int c = lane; c < 64; c += 32) {
        float mx = -3.0e38f;
#pragma unroll
        for (int k = 0; k < 16; ++k) mx = fmaxf(mx, logitS[wv][k][c]);
        float s = 0.f, r = 0.f;
        float e[16];
#pragma unroll
        for (int k = 0; k < 16; ++k) { e[k] = __expf(logitS[wv][k][c] - mx); s += e[k]; }
        float inv = 1.f / s;
#pragma unroll
        for (int k = 0; k < 16; ++k)
            r += e[k] * inv * (float)reinterpret_cast<const __bf16*>(&Af32[wv][k][0])[c];
        res[((size_t)b * N + n) * 64 + c] = r;
    }
}

// ---------------------------------------------------------------------------
// Brute-force KNN (k<=16, squared L2), one thread per query, LDS-staged source.
// ---------------------------------------------------------------------------
__global__ __launch_bounds__(128)
void knn_kernel(const float* __restrict__ qxyz, const float* __restrict__ sxyz,
                int* __restrict__ out, int Nq, int Ns, int kk)
{
    __shared__ float sx[128][3];
    const int b  = blockIdx.y;
    const int qi = blockIdx.x * 128 + threadIdx.x;
    const int qv = (qi < Nq) ? qi : (Nq - 1);

    const float px = qxyz[((size_t)b * Nq + qv) * 3 + 0];
    const float py = qxyz[((size_t)b * Nq + qv) * 3 + 1];
    const float pz = qxyz[((size_t)b * Nq + qv) * 3 + 2];

    float bd[16]; int bi[16];
#pragma unroll
    for (int i = 0; i < 16; ++i) { bd[i] = 3.0e38f; bi[i] = 0; }

    for (int base = 0; base < Ns; base += 128) {
        int j = base + threadIdx.x;
        if (j < Ns) {
            sx[threadIdx.x][0] = sxyz[((size_t)b * Ns + j) * 3 + 0];
            sx[threadIdx.x][1] = sxyz[((size_t)b * Ns + j) * 3 + 1];
            sx[threadIdx.x][2] = sxyz[((size_t)b * Ns + j) * 3 + 2];
        }
        __syncthreads();
        int cnt = Ns - base; if (cnt > 128) cnt = 128;
        for (int t = 0; t < cnt; ++t) {
            float dx = px - sx[t][0], dy = py - sx[t][1], dz = pz - sx[t][2];
            float d = dx * dx + dy * dy + dz * dz;
            if (d < bd[15]) {          // branchless stable insertion (asc, ties keep old)
                int jj = base + t;
#pragma unroll
                for (int i = 15; i > 0; --i) {
                    bool up = d < bd[i - 1];
                    float nb = up ? bd[i - 1] : ((d < bd[i]) ? d : bd[i]);
                    int   ni = up ? bi[i - 1] : ((d < bd[i]) ? jj : bi[i]);
                    bd[i] = nb; bi[i] = ni;
                }
                if (d < bd[0]) { bd[0] = d; bi[0] = jj; }
            }
        }
        __syncthreads();
    }
    if (qi < Nq)
        for (int i = 0; i < kk; ++i) out[((size_t)b * Nq + qi) * 16 + i] = bi[i];
}

// ---------------------------------------------------------------------------
// Farthest point sampling: one workgroup per batch, dist array in LDS,
// sequential m iterations with argmax tree reduction.
// ---------------------------------------------------------------------------
__global__ __launch_bounds__(256)
void fps_kernel(const float* __restrict__ xyz, int* __restrict__ fidx, int N, int m)
{
    __shared__ float dist[1024];
    __shared__ float rv[256];
    __shared__ int   ri[256];
    __shared__ float cx, cy, cz;
    __shared__ int   far;

    const int b = blockIdx.x;
    const int tid = threadIdx.x;
    const float* xb = xyz + (size_t)b * N * 3;

    for (int j = tid; j < N; j += 256) dist[j] = 1.0e10f;
    if (tid == 0) far = 0;
    __syncthreads();

    for (int it = 0; it < m; ++it) {
        if (tid == 0) {
            fidx[b * m + it] = far;                  // scan emits pre-update carry
            cx = xb[far * 3]; cy = xb[far * 3 + 1]; cz = xb[far * 3 + 2];
        }
        __syncthreads();
        float bv = -1.f; int bj = 0;
        for (int j = tid; j < N; j += 256) {
            float dx = xb[j * 3] - cx, dy = xb[j * 3 + 1] - cy, dz = xb[j * 3 + 2] - cz;
            float d = dx * dx + dy * dy + dz * dz;
            float nd = fminf(dist[j], d);
            dist[j] = nd;
            if (nd > bv) { bv = nd; bj = j; }
        }
        rv[tid] = bv; ri[tid] = bj;
        __syncthreads();
        for (int s = 128; s > 0; s >>= 1) {
            if (tid < s) {
                if (rv[tid + s] > rv[tid] ||
                    (rv[tid + s] == rv[tid] && ri[tid + s] < ri[tid])) {
                    rv[tid] = rv[tid + s]; ri[tid] = ri[tid + s];
                }
            }
            __syncthreads();
        }
        if (tid == 0) far = ri[0];
        __syncthreads();
    }
}

// ---------------------------------------------------------------------------
// Small utility kernels
// ---------------------------------------------------------------------------
__global__ void extract_xyz_kernel(const float* __restrict__ x, float* __restrict__ xyz, int N)
{
    int b = blockIdx.y;
    int t = blockIdx.x * 256 + threadIdx.x;
    if (t < N * 3) {
        int n = t / 3, d = t % 3;
        xyz[((size_t)b * N + n) * 3 + d] = x[((size_t)b * N + n) * 6 + d];
    }
}

__global__ void gather_rows_kernel(const float* __restrict__ src, const int* __restrict__ idx,
                                   float* __restrict__ dst, int Nsrc, int R, int C)
{
    int b = blockIdx.y;
    int t = blockIdx.x * 256 + threadIdx.x;
    if (t < R * C) {
        int r = t / C, c = t % C;
        int j = idx[(size_t)b * R + r];
        dst[((size_t)b * R + r) * C + c] = src[((size_t)b * Nsrc + j) * C + c];
    }
}

__global__ void maxpool16_kernel(const float* __restrict__ h, float* __restrict__ out,
                                 int m, int C)
{
    int b = blockIdx.y;
    int t = blockIdx.x * 256 + threadIdx.x;
    if (t < m * C) {
        int i = t / C, c = t % C;
        const float* p = h + (((size_t)b * m + i) * 16) * C + c;
        float v = -3.0e38f;
        for (int k = 0; k < 16; ++k) v = fmaxf(v, p[(size_t)k * C]);
        out[((size_t)b * m + i) * C + c] = v;
    }
}

__global__ void mean_kernel(const float* __restrict__ f, float* __restrict__ g,
                            int n, int C, int total)
{
    int t = blockIdx.x * 256 + threadIdx.x;
    if (t < total) {
        int b = t / C, c = t % C;
        float s = 0.f;
        for (int k = 0; k < n; ++k) s += f[((size_t)b * n + k) * C + c];
        g[t] = s / (float)n;
    }
}

// ---------------------------------------------------------------------------
// Host orchestration
// ---------------------------------------------------------------------------
struct PTP {
    const float *fc1w, *fc1b, *wq, *wk, *wv;
    const float *d0w, *d0b, *d1w, *d1b, *g0w, *g0b, *g1w, *g1b;
    const float *fc2w, *fc2b;
};

extern "C" void kernel_launch(void* const* d_in, const int* in_sizes, int n_in,
                              void* d_out, int out_size, void* d_ws, size_t ws_size,
                              hipStream_t stream)
{
    (void)in_sizes; (void)n_in; (void)out_size; (void)ws_size;
    const int B = 64;
    const float* x = (const float*)d_in[0];

    // ---- parse params in setup_inputs() insertion order ----
    int pi = 1;
    auto nxt = [&]() { return (const float*)d_in[pi++]; };
    auto readPT = [&]() {
        PTP p;
        p.fc1w = nxt(); p.fc1b = nxt();
        p.wq = nxt(); p.wk = nxt(); p.wv = nxt();
        p.d0w = nxt(); p.d0b = nxt();
        p.d1w = nxt(); p.d1b = nxt();
        p.g0w = nxt(); p.g0b = nxt();
        p.g1w = nxt(); p.g1b = nxt();
        p.fc2w = nxt(); p.fc2b = nxt();
        return p;
    };
    const float* m10w = nxt(); const float* m10b = nxt();
    const float* m11w = nxt(); const float* m11b = nxt();
    PTP trI = readPT();
    const float* dw[4]; const float* db[4]; PTP tr[4];
    for (int i = 0; i < 4; ++i) { dw[i] = nxt(); db[i] = nxt(); tr[i] = readPT(); }
    const float* m20w = nxt(); const float* m20b = nxt();
    const float* m21w = nxt(); const float* m21b = nxt();
    const float* m22w = nxt(); const float* m22b = nxt();

    // ---- workspace bump allocator ----
    char* ws = (char*)d_ws;
    size_t off = 0;
    auto alloc = [&](size_t bytes) -> void* {
        void* p = ws + off;
        off = (off + bytes + 255) & ~(size_t)255;
        return p;
    };
    float* xyz0   = (float*)alloc((size_t)B * 1024 * 3 * 4);
    float* xyz1   = (float*)alloc((size_t)B * 1024 * 3 * 4);
    float* fA     = (float*)alloc((size_t)B * 32768 * 4);       // max B*N*C
    float* fB     = (float*)alloc((size_t)B * 32768 * 4);
    float* x1buf  = (float*)alloc((size_t)B * 1024 * 64 * 4);
    float* qbuf   = (float*)alloc((size_t)B * 1024 * 64 * 4);
    float* kbuf   = (float*)alloc((size_t)B * 1024 * 64 * 4);
    float* vbuf   = (float*)alloc((size_t)B * 1024 * 64 * 4);
    float* resbuf = (float*)alloc((size_t)B * 1024 * 64 * 4);
    int*   idxbuf = (int*)  alloc((size_t)B * 1024 * 16 * 4);
    int*   fidxbf = (int*)  alloc((size_t)B * 256 * 4);
    float* grpbuf = (float*)alloc((size_t)B * 131072 * 4);      // down0 grouped
    float* hbuf   = (float*)alloc((size_t)B * 262144 * 4);      // down0 mlp out
    float* gbuf   = (float*)alloc((size_t)B * 512 * 4);
    float* h1buf  = (float*)alloc((size_t)B * 256 * 4);
    float* h2buf  = (float*)alloc((size_t)B * 64 * 4);

    auto gemm = [&](const float* A, const float* W, const float* bias,
                    const float* resid, float* C, int M, int N, int K, int relu) {
        dim3 g((M + 63) / 64, (N + 15) / 16);
        gemm_bf16_kernel<<<g, 128, 0, stream>>>(A, W, bias, resid, C, M, N, K, relu);
    };

    auto run_pt = [&](const PTP& p, const float* xyz, const float* fin, float* fout,
                      int N, int dp) {
        int kk = (N < 16) ? N : 16;
        { dim3 g((N + 127) / 128, B);
          knn_kernel<<<g, 128, 0, stream>>>(xyz, xyz, idxbuf, N, N, kk); }
        gemm(fin,   p.fc1w, p.fc1b, nullptr, x1buf, B * N, 64, dp, 0);
        gemm(x1buf, p.wq,   nullptr, nullptr, qbuf, B * N, 64, 64, 0);
        gemm(x1buf, p.wk,   nullptr, nullptr, kbuf, B * N, 64, 64, 0);
        gemm(x1buf, p.wv,   nullptr, nullptr, vbuf, B * N, 64, 64, 0);
        { dim3 g((N + 3) / 4, B);
          pt_attn_kernel<<<g, 128, 0, stream>>>(xyz, idxbuf, qbuf, kbuf, vbuf,
                                                p.d0w, p.d0b, p.d1w, p.d1b,
                                                p.g0w, p.g0b, p.g1w, p.g1b,
                                                resbuf, N, kk); }
        gemm(resbuf, p.fc2w, p.fc2b, fin, fout, B * N, dp, 64, 0);
    };

    // ---- stem ----
    { dim3 g((1024 * 3 + 255) / 256, B);
      extract_xyz_kernel<<<g, 256, 0, stream>>>(x, xyz0, 1024); }
    gemm(x,  m10w, m10b, nullptr, fB, B * 1024, 32, 6, 1);   // relu(linear)
    gemm(fB, m11w, m11b, nullptr, fA, B * 1024, 32, 32, 0);  // linear

    float* xyzCur = xyz0; float* xyzNxt = xyz1;
    float* fCur = fA;     float* fNxt = fB;

    run_pt(trI, xyzCur, fCur, fNxt, 1024, 32);
    { float* t = fCur; fCur = fNxt; fNxt = t; }

    int N = 1024;
    int cin = 32;
    for (int i = 0; i < 4; ++i) {
        int m = N / 4;
        int cout = cin * 2;
        // FPS + gather new xyz
        fps_kernel<<<B, 256, 0, stream>>>(xyzCur, fidxbf, N, m);
        { dim3 g((m * 3 + 255) / 256, B);
          gather_rows_kernel<<<g, 256, 0, stream>>>(xyzCur, fidxbf, xyzNxt, N, m, 3); }
        // KNN new->old, gather features, MLP, max-pool
        { dim3 g((m + 127) / 128, B);
          knn_kernel<<<g, 128, 0, stream>>>(xyzNxt, xyzCur, idxbuf, m, N, 16); }
        { dim3 g((m * 16 * cin + 255) / 256, B);
          gather_rows_kernel<<<g, 256, 0, stream>>>(fCur, idxbuf, grpbuf, N, m * 16, cin); }
        gemm(grpbuf, dw[i], db[i], nullptr, hbuf, B * m * 16, cout, cin, 1);
        { dim3 g((m * cout + 255) / 256, B);
          maxpool16_kernel<<<g, 256, 0, stream>>>(hbuf, fNxt, m, cout); }
        { float* t = fCur; fCur = fNxt; fNxt = t; }
        { float* t = xyzCur; xyzCur = xyzNxt; xyzNxt = t; }
        // transformer block at the new resolution
        run_pt(tr[i], xyzCur, fCur, fNxt, m, cout);
        { float* t = fCur; fCur = fNxt; fNxt = t; }
        N = m; cin = cout;
    }

    // ---- head: mean over 4 points, 512->256->64->40 ----
    { int total = B * 512;
      mean_kernel<<<(total + 255) / 256, 256, 0, stream>>>(fCur, gbuf, 4, 512, total); }
    gemm(gbuf,  m20w, m20b, nullptr, h1buf, B, 256, 512, 1);
    gemm(h1buf, m21w, m21b, nullptr, h2buf, B, 64, 256, 1);
    gemm(h2buf, m22w, m22b, nullptr, (float*)d_out, B, 40, 64, 0);
}
